// NodeUpdateBlock_6253472383850
// MI455X (gfx1250) — compile-verified, matrix operand order checked by reading, not run on hardware
//
#include <hip/hip_runtime.h>

typedef __attribute__((ext_vector_type(16))) _Float16 v16h;
typedef __attribute__((ext_vector_type(8)))  float    v8f;
typedef __attribute__((ext_vector_type(4)))  int      v4i;

#define DN 128
#define DE 64
#define NH 8
#define CD 16

#define AS1 __attribute__((address_space(1)))
#define AS3 __attribute__((address_space(3)))

#if defined(__has_builtin)
#  if __has_builtin(__builtin_amdgcn_global_load_async_to_lds_b128)
#    define HAVE_ASYNC_LDS 1
#  endif
#endif
#ifndef HAVE_ASYNC_LDS
#  define HAVE_ASYNC_LDS 0
#endif

__device__ __forceinline__ void async_wait_all() {
#if __has_builtin(__builtin_amdgcn_s_wait_asynccnt)
  __builtin_amdgcn_s_wait_asynccnt(0);
#else
  asm volatile("s_wait_asynccnt 0x0" ::: "memory");
#endif
}

#if HAVE_ASYNC_LDS
// async-copy 16 contiguous bytes global -> LDS (ASYNCcnt tracked)
__device__ __forceinline__ void async_copy16(const float* gsrc, float* lds_dst) {
  __builtin_amdgcn_global_load_async_to_lds_b128(
      (AS1 v4i*)(void*)gsrc, (AS3 v4i*)(void*)lds_dst, 0, 0);
}
#endif

// K-permutation within a 32-wide K tile for the 16-bit WMMA operand layout:
// lanes 0-15: j=0..7 -> K 0..7, j=8..15 -> K 16..23 ; lanes 16-31: +8
__device__ __forceinline__ int kperm(int j, int lane) {
  return (j & 7) + ((j >> 3) << 4) + ((lane >> 4) << 3);
}

// ---------------------------------------------------------------------------
// Pack row-major f32 W[K][Ncols] into per-fragment f16 layout:
// frag(kt,nt) = 32 lanes x 16 halves, lane's 16 halves contiguous (32B loads).
// ---------------------------------------------------------------------------
__global__ void pack_w(const float* __restrict__ W, _Float16* __restrict__ dst,
                       int K, int Ncols) {
  int tid = blockIdx.x * blockDim.x + threadIdx.x;
  int total = K * Ncols;
  if (tid >= total) return;
  int NT   = Ncols >> 4;
  int frag = tid >> 9;
  int r    = tid & 511;
  int lane = r >> 4;
  int j    = r & 15;
  int kt = frag / NT, nt = frag % NT;
  int k   = kt * 32 + kperm(j, lane);
  int col = nt * 16 + (lane & 15);
  dst[tid] = (_Float16)W[k * Ncols + col];
}

__global__ void zero_f32(float* __restrict__ p, int n) {
  int i = blockIdx.x * blockDim.x + threadIdx.x;
  if (i < n) p[i] = 0.0f;
}

// ---------------------------------------------------------------------------
// Node pass: q/k/v/skip = node_state @ W + b via WMMA.
// Block = 16 node rows, 4 waves: wave w owns weight w (all 8 N-tiles).
// Outputs staged in LDS (branch-free) then flushed with coalesced b128 stores.
// ---------------------------------------------------------------------------
__global__ void node_qkvs(const float* __restrict__ ns,
                          const _Float16* __restrict__ pWq, const _Float16* __restrict__ pWk,
                          const _Float16* __restrict__ pWv, const _Float16* __restrict__ pWs,
                          const float* __restrict__ bq, const float* __restrict__ bk,
                          const float* __restrict__ bv, const float* __restrict__ bs,
                          _Float16* __restrict__ q16, _Float16* __restrict__ k16,
                          _Float16* __restrict__ v16, float* __restrict__ xr, int N) {
  __shared__ __align__(16) float    atf[16 * DN];        // input tile (f32)
  __shared__ __align__(16) _Float16 st16[3 * 16 * DN];   // q/k/v staging
  __shared__ __align__(16) float    sxr[16 * DN];        // x_r staging
  int tid = threadIdx.x;
  int n0  = blockIdx.x * 16;
  bool full = (n0 + 16 <= N);

#if HAVE_ASYNC_LDS
  if (full) {
    for (int c = tid; c < 512; c += 128)          // 8KB tile, contiguous
      async_copy16(ns + n0 * DN + c * 4, atf + c * 4);
    async_wait_all();
  } else
#endif
  {
    for (int i = tid; i < 16 * DN; i += 128) {
      int g = n0 + i / DN;
      atf[i] = (g < N) ? ns[g * DN + (i % DN)] : 0.0f;
    }
  }
  __syncthreads();

  int wv = tid >> 5, lane = tid & 31;
  const _Float16* pW; const float* bias;
  if      (wv == 0) { pW = pWq; bias = bq; }
  else if (wv == 1) { pW = pWk; bias = bk; }
  else if (wv == 2) { pW = pWv; bias = bv; }
  else              { pW = pWs; bias = bs; }

  int m = lane & 15, lh = (lane >> 4) << 3;
  int colL = lane & 15, rowHi = (lane >> 4) << 3;
  v16h af[4];
#pragma unroll
  for (int kt = 0; kt < 4; ++kt)
#pragma unroll
    for (int j = 0; j < 16; ++j)
      af[kt][j] = (_Float16)atf[m * DN + kt * 32 + (j & 7) + ((j >> 3) << 4) + lh];

  _Float16* st = st16 + wv * (16 * DN);   // waves 0-2; wave 3 uses sxr
  for (int nt = 0; nt < 8; ++nt) {
    v8f acc = {};
#pragma unroll
    for (int kt = 0; kt < 4; ++kt) {
      v16h b = *(const v16h*)(pW + ((kt * 8 + nt) * 512 + lane * 16));
      acc = __builtin_amdgcn_wmma_f32_16x16x32_f16(false, af[kt], false, b,
                                                   (short)0, acc, false, false);
    }
    int col = nt * 16 + colL;
    float bb = bias[col];
    if (wv < 3) {
#pragma unroll
      for (int r = 0; r < 8; ++r) st[(r + rowHi) * DN + col] = (_Float16)(acc[r] + bb);
    } else {
#pragma unroll
      for (int r = 0; r < 8; ++r) sxr[(r + rowHi) * DN + col] = acc[r] + bb;
    }
  }
  __syncthreads();

  // coalesced flush: q/k/v tiles (256 x 16B each), x_r (512 x 16B)
#pragma unroll
  for (int w = 0; w < 3; ++w) {
    _Float16* dstb = (w == 0) ? q16 : ((w == 1) ? k16 : v16);
    for (int u = tid; u < 256; u += 128) {
      int row = u >> 4, col = (u & 15) << 3;
      int g = n0 + row;
      if (g < N)
        *(uint4*)(dstb + g * DN + col) = *(const uint4*)(st16 + w * (16 * DN) + row * DN + col);
    }
  }
  for (int u = tid; u < 512; u += 128) {
    int row = u >> 5, col = (u & 31) << 2;
    int g = n0 + row;
    if (g < N)
      *(uint4*)(xr + g * DN + col) = *(const uint4*)(sxr + row * DN + col);
  }
}

// ---------------------------------------------------------------------------
// Fused edge pass: edge_attr = es@W_edge+b ; e = edge_attr@We (both WMMA,
// never touch HBM); alpha = <q[dst], k[src]+e>/4; accumulate unnormalized
// softmax numerator (d_out) and denominator (denom) via f32 atomics; both
// accumulators are L2-resident (25.6MB + 1.6MB << 192MB L2).
// ---------------------------------------------------------------------------
__global__ void edge_pass(const float* __restrict__ es_g, const int* __restrict__ eidx,
                          const float* __restrict__ b_edge,
                          const _Float16* __restrict__ pWedge, const _Float16* __restrict__ pWe,
                          const _Float16* __restrict__ q16, const _Float16* __restrict__ k16,
                          const _Float16* __restrict__ v16,
                          float* __restrict__ denom, float* __restrict__ out_acc, int E) {
  __shared__ __align__(16) float    etf[16 * DE];   // edge tile (f32)
  __shared__ __align__(16) _Float16 ea[16 * DE];    // edge_attr (f16)
  __shared__ __align__(16) float    el[16 * DN];    // e tile (f32)
  __shared__ int ssrc[16], sdst[16];

  int tid = threadIdx.x;
  int eb  = blockIdx.x * 16;
  bool full = (eb + 16 <= E);

#if HAVE_ASYNC_LDS
  if (full) {
    for (int c = tid; c < 256; c += 128)          // 4KB tile, contiguous
      async_copy16(es_g + eb * DE + c * 4, etf + c * 4);
    async_wait_all();
  } else
#endif
  {
    for (int i = tid; i < 16 * DE; i += 128) {
      int g = eb + i / DE;
      etf[i] = (g < E) ? es_g[g * DE + (i % DE)] : 0.0f;
    }
  }
  if (tid < 16) {
    int g = eb + tid;
    ssrc[tid] = (g < E) ? eidx[g]     : -1;
    sdst[tid] = (g < E) ? eidx[E + g] : -1;
    int gn = g + 16;                              // prefetch next tile
    if (gn < E) __builtin_prefetch(&es_g[gn * DE], 0, 3);
  }
  __syncthreads();

  int wv = tid >> 5, lane = tid & 31;
  int m = lane & 15, lh = (lane >> 4) << 3;
  int colL = lane & 15, rowHi = (lane >> 4) << 3;

  // stage 1: edge_attr (16x64) = et @ W_edge + b ; ntile = wave id
  {
    v8f acc = {};
#pragma unroll
    for (int kt = 0; kt < 2; ++kt) {
      v16h a;
#pragma unroll
      for (int j = 0; j < 16; ++j)
        a[j] = (_Float16)etf[m * DE + kt * 32 + (j & 7) + ((j >> 3) << 4) + lh];
      v16h b = *(const v16h*)(pWedge + ((kt * 4 + wv) * 512 + lane * 16));
      acc = __builtin_amdgcn_wmma_f32_16x16x32_f16(false, a, false, b,
                                                   (short)0, acc, false, false);
    }
    int col = wv * 16 + colL;
    float bb = b_edge[col];
#pragma unroll
    for (int r = 0; r < 8; ++r) ea[(r + rowHi) * DE + col] = (_Float16)(acc[r] + bb);
  }
  __syncthreads();

  // stage 2: e (16x128) = ea @ We ; wave handles ntiles {2w, 2w+1}
#pragma unroll
  for (int t = 0; t < 2; ++t) {
    int nt = wv * 2 + t;
    v8f acc = {};
#pragma unroll
    for (int kt = 0; kt < 2; ++kt) {
      v16h a;
#pragma unroll
      for (int j = 0; j < 16; ++j)
        a[j] = ea[m * DE + kt * 32 + (j & 7) + ((j >> 3) << 4) + lh];
      v16h b = *(const v16h*)(pWe + ((kt * 8 + nt) * 512 + lane * 16));
      acc = __builtin_amdgcn_wmma_f32_16x16x32_f16(false, a, false, b,
                                                   (short)0, acc, false, false);
    }
    int col = nt * 16 + colL;
#pragma unroll
    for (int r = 0; r < 8; ++r) el[(r + rowHi) * DN + col] = acc[r];
  }
  __syncthreads();

  // stage 3: per (edge, head) attention + scatter-accumulate
  int m2 = tid >> 3, h = tid & 7;
  int s = ssrc[m2], d = sdst[m2];
  if (s >= 0) {
    v16h qv = *(const v16h*)(q16 + d * DN + h * CD);   // 32B vector gathers
    v16h kv = *(const v16h*)(k16 + s * DN + h * CD);
    v16h vv = *(const v16h*)(v16 + s * DN + h * CD);
    const float* ee = el + m2 * DN + h * CD;
    float alpha = 0.0f;
#pragma unroll
    for (int c = 0; c < CD; ++c) alpha += (float)qv[c] * ((float)kv[c] + ee[c]);
    float ex = __expf(alpha * 0.25f);                  // 1/sqrt(C), C=16
    atomicAdd(&denom[d * NH + h], ex);
    float* oa = out_acc + d * DN + h * CD;
#pragma unroll
    for (int c = 0; c < CD; ++c) atomicAdd(&oa[c], ((float)vv[c] + ee[c]) * ex);
  }
}

// ---------------------------------------------------------------------------
// Final node pass: normalize softmax, beta-gated skip, LayerNorm, res + relu.
// ---------------------------------------------------------------------------
__global__ void node_final(const float* __restrict__ ns, const float* __restrict__ xr,
                           const float* __restrict__ denom, const float* __restrict__ Wbeta,
                           const float* __restrict__ lng, const float* __restrict__ lnb,
                           float* __restrict__ out, int N) {
  __shared__ float red[DN];
  int n = blockIdx.x, t = threadIdx.x;
  if (n >= N) return;

  float agg = out[n * DN + t];
  float den = denom[n * NH + (t >> 4)];
  float o   = (den > 0.0f) ? agg / den : 0.0f;
  float x   = xr[n * DN + t];

  red[t] = o * Wbeta[t] + x * Wbeta[DN + t] + (o - x) * Wbeta[2 * DN + t];
  __syncthreads();
  for (int off = 64; off > 0; off >>= 1) { if (t < off) red[t] += red[t + off]; __syncthreads(); }
  float beta = 1.0f / (1.0f + __expf(-red[0]));
  __syncthreads();

  float g = beta * x + (1.0f - beta) * o;

  red[t] = g; __syncthreads();
  for (int off = 64; off > 0; off >>= 1) { if (t < off) red[t] += red[t + off]; __syncthreads(); }
  float mu = red[0] * (1.0f / DN);
  __syncthreads();
  float dmu = g - mu;
  red[t] = dmu * dmu; __syncthreads();
  for (int off = 64; off > 0; off >>= 1) { if (t < off) red[t] += red[t + off]; __syncthreads(); }
  float var = red[0] * (1.0f / DN);

  float nrm = dmu * rsqrtf(var + 1e-5f) * lng[t] + lnb[t];
  out[n * DN + t] = ns[n * DN + t] + fmaxf(nrm, 0.0f);
}

// ---------------------------------------------------------------------------
extern "C" void kernel_launch(void* const* d_in, const int* in_sizes, int n_in,
                              void* d_out, int out_size, void* d_ws, size_t ws_size,
                              hipStream_t stream) {
  const float* node_state = (const float*)d_in[0];
  const int*   edge_index = (const int*)d_in[1];
  const float* edge_state = (const float*)d_in[2];
  const float* W_edge     = (const float*)d_in[3];
  const float* b_edge     = (const float*)d_in[4];
  const float* Wq         = (const float*)d_in[5];
  const float* bq         = (const float*)d_in[6];
  const float* Wk         = (const float*)d_in[7];
  const float* bk         = (const float*)d_in[8];
  const float* Wv         = (const float*)d_in[9];
  const float* bv         = (const float*)d_in[10];
  const float* We         = (const float*)d_in[11];
  const float* Wskip      = (const float*)d_in[12];
  const float* bskip      = (const float*)d_in[13];
  const float* Wbeta      = (const float*)d_in[14];
  const float* lng        = (const float*)d_in[15];
  const float* lnb        = (const float*)d_in[16];

  int N = in_sizes[0] / DN;
  int E = in_sizes[1] / 2;
  float* out = (float*)d_out;

  _Float16* pWq    = (_Float16*)d_ws;
  _Float16* pWk    = pWq    + DN * DN;
  _Float16* pWv    = pWk    + DN * DN;
  _Float16* pWs    = pWv    + DN * DN;
  _Float16* pWedge = pWs    + DN * DN;
  _Float16* pWe    = pWedge + DE * DE;
  _Float16* q16    = pWe    + DE * DN;
  _Float16* k16    = q16 + (size_t)N * DN;
  _Float16* v16    = k16 + (size_t)N * DN;
  float*    xr     = (float*)(v16 + (size_t)N * DN);
  float*    denom  = xr + (size_t)N * DN;

  zero_f32<<<(N * DN + 255) / 256, 256, 0, stream>>>(out, N * DN);
  zero_f32<<<(N * NH + 255) / 256, 256, 0, stream>>>(denom, N * NH);

  pack_w<<<(DN * DN + 255) / 256, 256, 0, stream>>>(Wq,     pWq,    DN, DN);
  pack_w<<<(DN * DN + 255) / 256, 256, 0, stream>>>(Wk,     pWk,    DN, DN);
  pack_w<<<(DN * DN + 255) / 256, 256, 0, stream>>>(Wv,     pWv,    DN, DN);
  pack_w<<<(DN * DN + 255) / 256, 256, 0, stream>>>(Wskip,  pWs,    DN, DN);
  pack_w<<<(DE * DE + 255) / 256, 256, 0, stream>>>(W_edge, pWedge, DE, DE);
  pack_w<<<(DE * DN + 255) / 256, 256, 0, stream>>>(We,     pWe,    DE, DN);

  node_qkvs<<<(N + 15) / 16, 128, 0, stream>>>(node_state, pWq, pWk, pWv, pWs,
                                               bq, bk, bv, bskip,
                                               q16, k16, v16, xr, N);

  edge_pass<<<(E + 15) / 16, 128, 0, stream>>>(edge_state, edge_index, b_edge,
                                               pWedge, pWe, q16, k16, v16,
                                               denom, out, E);

  node_final<<<N, 128, 0, stream>>>(node_state, xr, denom, Wbeta, lng, lnb, out, N);
}
